// Attention_51135880626809
// MI455X (gfx1250) — matrix operand order repaired and checked by reading, not compiled, not run on hardware
//
#include <hip/hip_runtime.h>
#include <hip/hip_bf16.h>

// ---------------------------------------------------------------------------
// MI455X (gfx1250) fused LN + QKV projection + flash attention + out proj.
// f16 WMMA (v_wmma_f32_16x16x32_f16) with f32 accumulation everywhere.
// GEMM: LDS-staged double-buffered weight tiles (async-to-LDS DMA),
// 32x64 per wave. Attention: online softmax, row-sums via P@ones WMMA,
// softmax scale folded into the Q projection. Workspace use: ~56 MiB.
// ---------------------------------------------------------------------------

typedef _Float16 h16v __attribute__((ext_vector_type(16)));
typedef _Float16 h8v  __attribute__((ext_vector_type(8)));
typedef _Float16 h4v  __attribute__((ext_vector_type(4)));
typedef float    f8v  __attribute__((ext_vector_type(8)));
typedef float    f4v  __attribute__((ext_vector_type(4)));
typedef int      i4v  __attribute__((ext_vector_type(4)));
typedef __attribute__((address_space(1))) i4v i4v_g;   // global int4
typedef __attribute__((address_space(3))) i4v i4v_l;   // LDS int4

#define DIM   1024
#define NSEQ  2048
#define BATCH 2
#define HEADS 16
#define HD    64
#define ROWS  (BATCH * NSEQ)          // 4096
#define ATT_SCALE 0.125f              // 64^-0.5
#define LOG2E 1.4426950408889634f
#define QSCALE (ATT_SCALE * LOG2E)    // folded into Q projection; exp2 in loop

#if defined(__has_builtin)
#if __has_builtin(__builtin_amdgcn_global_load_async_to_lds_b128) && \
    __has_builtin(__builtin_amdgcn_s_wait_asynccnt)
#define HAVE_ASYNC_LDS 1
#endif
#endif

__device__ __forceinline__ h16v h16_combine(h8v lo, h8v hi) {
    h16v r;
#pragma unroll
    for (int i = 0; i < 8; ++i) { r[i] = lo[i]; r[i + 8] = hi[i]; }
    return r;
}

__device__ __forceinline__ f8v wmma_f16(h16v a, h16v b, f8v c) {
    return __builtin_amdgcn_wmma_f32_16x16x32_f16(false, a, false, b,
                                                  (short)0, c, false, false);
}

// A fragment: row = fixed per lane, K = half*16 + e
__device__ __forceinline__ h16v load_afrag(const _Float16* row, int k0, int half) {
    return h16_combine(*(const h8v*)(row + k0 + half * 16),
                       *(const h8v*)(row + k0 + half * 16 + 8));
}

// Stage 16 bytes global -> LDS via gfx1250 async-to-LDS DMA (ASYNCcnt-tracked,
// no VGPR round trip); plain copy fallback.
__device__ __forceinline__ void stage16(_Float16* lds_dst, const _Float16* gsrc) {
#if defined(HAVE_ASYNC_LDS)
    __builtin_amdgcn_global_load_async_to_lds_b128(
        (i4v_g*)gsrc, (i4v_l*)lds_dst, 0, 0);
#else
    *(h8v*)lds_dst = *(const h8v*)gsrc;
#endif
}

__device__ __forceinline__ void stage_wait() {
#if defined(HAVE_ASYNC_LDS)
    __builtin_amdgcn_s_wait_asynccnt(0);
#endif
}

// Compiler-only reordering fence: same-wave LDS ops execute in order (ISA
// §7.3, DScnt in-order), so a wave-private LDS store->load needs no barrier.
__device__ __forceinline__ void wave_lds_fence() {
    asm volatile("" ::: "memory");
}

// ---------------------------------------------------------------------------
// LayerNorm + f32 -> f16.  One 256-thread block per row (1024 elems).
// blockIdx.y selects which of q/k/v.
// ---------------------------------------------------------------------------
__global__ void ln_kernel(const float* __restrict__ q, const float* __restrict__ k,
                          const float* __restrict__ v,
                          const float* __restrict__ gq, const float* __restrict__ bq,
                          const float* __restrict__ gk, const float* __restrict__ bk,
                          const float* __restrict__ gv, const float* __restrict__ bv,
                          _Float16* __restrict__ oq, _Float16* __restrict__ ok,
                          _Float16* __restrict__ ov) {
    int row = blockIdx.x;
    int which = blockIdx.y;
    const float *x, *g, *b;
    _Float16* o;
    if (which == 0)      { x = q; g = gq; b = bq; o = oq; }
    else if (which == 1) { x = k; g = gk; b = bk; o = ok; }
    else                 { x = v; g = gv; b = bv; o = ov; }

    const float* xr = x + (size_t)row * DIM;
    int t = threadIdx.x;
    f4v xv = *(const f4v*)(xr + t * 4);
    float s  = xv[0] + xv[1] + xv[2] + xv[3];
    float sq = xv[0]*xv[0] + xv[1]*xv[1] + xv[2]*xv[2] + xv[3]*xv[3];

    __shared__ float rs[256], rq[256];
    rs[t] = s; rq[t] = sq;
    __syncthreads();
#pragma unroll
    for (int off = 128; off > 0; off >>= 1) {
        if (t < off) { rs[t] += rs[t + off]; rq[t] += rq[t + off]; }
        __syncthreads();
    }
    float mu   = rs[0] * (1.0f / DIM);
    float var  = rq[0] * (1.0f / DIM) - mu * mu;
    float rstd = rsqrtf(var + 1e-5f);

    h4v y16;
#pragma unroll
    for (int i = 0; i < 4; ++i) {
        int c = t * 4 + i;
        y16[i] = (_Float16)((xv[i] - mu) * rstd * g[c] + b[c]);
    }
    *(h4v*)(o + (size_t)row * DIM + t * 4) = y16;
}

// ---------------------------------------------------------------------------
// Weight convert f32 -> f16, transposed via LDS tile (coalesced both ways):
// Wt[n][k] = W[k][n], so a WMMA B-fragment (one column, K-major) is contiguous.
// grid (DIM/32, DIM/32, 4), block (32,8).
// ---------------------------------------------------------------------------
__global__ void cvtw_kernel(const float* __restrict__ wq, const float* __restrict__ wk,
                            const float* __restrict__ wv, const float* __restrict__ wp,
                            _Float16* __restrict__ oq, _Float16* __restrict__ ok,
                            _Float16* __restrict__ ov, _Float16* __restrict__ op) {
    int which = blockIdx.z;
    const float* w; _Float16* o;
    if (which == 0)      { w = wq; o = oq; }
    else if (which == 1) { w = wk; o = ok; }
    else if (which == 2) { w = wv; o = ov; }
    else                 { w = wp; o = op; }

    __shared__ float tile[32][33];
    int k0 = blockIdx.x * 32;
    int n0 = blockIdx.y * 32;
    int tx = threadIdx.x, ty = threadIdx.y;
#pragma unroll
    for (int i = 0; i < 4; ++i) {
        int kk = ty + 8 * i;
        tile[kk][tx] = w[(size_t)(k0 + kk) * DIM + n0 + tx];   // coalesced read
    }
    __syncthreads();
#pragma unroll
    for (int i = 0; i < 4; ++i) {
        int nn = ty + 8 * i;
        o[(size_t)(n0 + nn) * DIM + k0 + tx] = (_Float16)tile[tx][nn]; // coalesced
    }
}

// ---------------------------------------------------------------------------
// C[M=4096, N=1024] = A16[M,1024] @ W (Wt is [N,K] f16) + bias.
// Block = 8 waves; wave w -> rows m0..m0+31 (2 M-tiles), cols n0..n0+63
// (4 N-tiles) => 8 WMMAs per 32-wide k-step.
// Weight tile (64 cols x 32 k = 4 KB) staged in LDS via async DMA,
// double buffered, shared by all 8 waves.
// MODE32: 0 none, 1 row-major [M,DIM], 2 head-major [B,H,N,hd]
// MODE16: 0 none, 1 head-major [B,H,N,hd], 2 d-major  [B,H,hd,N]
// SCALEQ: multiply the f16 output by QSCALE (for the Q projection).
// ---------------------------------------------------------------------------
template <int MODE32, int MODE16, bool SCALEQ>
__global__ __launch_bounds__(256, 1)
void gemm_kernel(const _Float16* __restrict__ A,
                 const _Float16* __restrict__ Wt,
                 const float* __restrict__ bias,
                 float* __restrict__ out32, _Float16* __restrict__ out16) {
    int lane = threadIdx.x & 31;
    int wave = threadIdx.x >> 5;
    int half = lane >> 4;
    int lrow = lane & 15;
    int m0 = blockIdx.x * 256 + wave * 32;
    int n0 = blockIdx.y * 64;

    __shared__ __align__(16) _Float16 bstage[2][64][32];   // 8 KB double buffer

    // cooperative B-tile stage: thread t copies 8 f16 of column (n0 + t/4)
    int sn = threadIdx.x >> 2;            // 0..63
    int sp = (threadIdx.x & 3) * 8;       // 0,8,16,24
    const _Float16* srcb = Wt + (size_t)(n0 + sn) * DIM + sp;

    f8v acc[2][4] = {};
    const _Float16* arow0 = A + (size_t)(m0 + lrow) * DIM;
    const _Float16* arow1 = A + (size_t)(m0 + 16 + lrow) * DIM;

    // prologue: stage k-tile 0, preload A fragments for k-tile 0
    stage16(&bstage[0][sn][sp], srcb);
    h16v af0n = load_afrag(arow0, 0, half);
    h16v af1n = load_afrag(arow1, 0, half);
    stage_wait();
    __syncthreads();

#pragma unroll 2
    for (int k0 = 0; k0 < DIM; k0 += 32) {
        int buf = (k0 >> 5) & 1;
        h16v af0 = af0n, af1 = af1n;
        if (k0 + 32 < DIM) {
            // overlap: stage next B tile + next A fragments during WMMAs
            stage16(&bstage[buf ^ 1][sn][sp], srcb + k0 + 32);
            af0n = load_afrag(arow0, k0 + 32, half);
            af1n = load_afrag(arow1, k0 + 32, half);
        }
        // B fragments from LDS: col = n0+16j+lrow, K = half*16 + e
        h16v bf[4];
#pragma unroll
        for (int j = 0; j < 4; ++j)
            bf[j] = h16_combine(*(const h8v*)&bstage[buf][16 * j + lrow][half * 16],
                                *(const h8v*)&bstage[buf][16 * j + lrow][half * 16 + 8]);
#pragma unroll
        for (int j = 0; j < 4; ++j) {
            acc[0][j] = wmma_f16(af0, bf[j], acc[0][j]);
            acc[1][j] = wmma_f16(af1, bf[j], acc[1][j]);
        }
        stage_wait();
        __syncthreads();
    }

#pragma unroll
    for (int i = 0; i < 2; ++i) {
#pragma unroll
        for (int j = 0; j < 4; ++j) {
            int c = n0 + 16 * j + lrow;      // output column (D: N = lane&15)
            float bv = bias[c];
            int h = c >> 6, d = c & 63;
#pragma unroll
            for (int r = 0; r < 8; ++r) {
                int m = m0 + 16 * i + r + 8 * half;  // row (D: M = r + 8*half)
                float val = acc[i][j][r] + bv;
                int bi = m / NSEQ, n = m - bi * NSEQ;
                if (MODE32 == 1)
                    out32[(size_t)m * DIM + c] = val;
                else if (MODE32 == 2)
                    out32[(((size_t)bi * HEADS + h) * NSEQ + n) * HD + d] = val;
                float v16 = SCALEQ ? val * QSCALE : val;
                if (MODE16 == 1)
                    out16[(((size_t)bi * HEADS + h) * NSEQ + n) * HD + d] = (_Float16)v16;
                else if (MODE16 == 2)
                    out16[(((size_t)bi * HEADS + h) * HD + d) * NSEQ + n] = (_Float16)v16;
            }
        }
    }
}

// ---------------------------------------------------------------------------
// Flash attention. Q (pre-scaled by QSCALE), K head-major f16 [B,H,N,64];
// Vt d-major f16 [B,H,64,N]. One wave owns 16 q-rows; streams keys in
// 32-wide steps with online softmax in exp2 domain (scale folded into Q).
// Row sums computed by an extra WMMA against an all-ones B matrix, so only
// the row-max butterfly uses cross-lane shuffles.
// ---------------------------------------------------------------------------
__global__ __launch_bounds__(256, 1)
void attn_kernel(const _Float16* __restrict__ Q,
                 const _Float16* __restrict__ K,
                 const _Float16* __restrict__ Vt,
                 _Float16* __restrict__ O) {
    int lane = threadIdx.x & 31;
    int wave = threadIdx.x >> 5;
    int half = lane >> 4;
    int lrow = lane & 15;
    int q0 = blockIdx.x * 128 + wave * 16;
    int h  = blockIdx.y;
    int b  = blockIdx.z;
    int bh = b * HEADS + h;

    // Q A-fragments (K dim = head_dim, split 0..31 / 32..63)
    const _Float16* qrow = Q + ((size_t)bh * NSEQ + q0 + lrow) * HD;
    h16v aq0 = h16_combine(*(const h8v*)(qrow +      half * 16),
                           *(const h8v*)(qrow + 8 + half * 16));
    h16v aq1 = h16_combine(*(const h8v*)(qrow + 32 + half * 16),
                           *(const h8v*)(qrow + 40 + half * 16));

    h16v ones;
#pragma unroll
    for (int i = 0; i < 16; ++i) ones[i] = (_Float16)1.0f;

    f8v acco[4] = {};
    float m_i[8], l_i[8];
#pragma unroll
    for (int r = 0; r < 8; ++r) { m_i[r] = -1e30f; l_i[r] = 0.0f; }

    __shared__ __align__(16) _Float16 plds[8][16 * 32];   // per-wave P tile

    const _Float16* kbase = K  + (size_t)bh * NSEQ * HD;
    const _Float16* vbase = Vt + (size_t)bh * HD * NSEQ;

    for (int kb = 0; kb < NSEQ; kb += 32) {
        // ---- issue all K and V fragment loads for this key block ----
        const _Float16* kr0 = kbase + (size_t)(kb + lrow) * HD + half * 16;
        const _Float16* kr1 = kbase + (size_t)(kb + 16 + lrow) * HD + half * 16;
        h16v kf00 = h16_combine(*(const h8v*)kr0,        *(const h8v*)(kr0 + 8));
        h16v kf01 = h16_combine(*(const h8v*)(kr0 + 32), *(const h8v*)(kr0 + 40));
        h16v kf10 = h16_combine(*(const h8v*)kr1,        *(const h8v*)(kr1 + 8));
        h16v kf11 = h16_combine(*(const h8v*)(kr1 + 32), *(const h8v*)(kr1 + 40));
        h16v vf[4];
#pragma unroll
        for (int j = 0; j < 4; ++j) {
            const _Float16* vr = vbase + (size_t)(16 * j + lrow) * NSEQ + kb + half * 16;
            vf[j] = h16_combine(*(const h8v*)vr, *(const h8v*)(vr + 8));
        }

        // ---- S = Q K^T (already in log2 units), two 16-key tiles ----
        f8v s0 = {}, s1 = {};
        s0 = wmma_f16(aq0, kf00, s0);
        s0 = wmma_f16(aq1, kf01, s0);
        s1 = wmma_f16(aq0, kf10, s1);
        s1 = wmma_f16(aq1, kf11, s1);

        // ---- online softmax: row max via 16-lane butterfly, p = exp2 ----
        float alpha[8];
#pragma unroll
        for (int r = 0; r < 8; ++r) {
            float mt = fmaxf(s0[r], s1[r]);
            mt = fmaxf(mt, __shfl_xor(mt, 1, 32));
            mt = fmaxf(mt, __shfl_xor(mt, 2, 32));
            mt = fmaxf(mt, __shfl_xor(mt, 4, 32));
            mt = fmaxf(mt, __shfl_xor(mt, 8, 32));
            float mn = fmaxf(m_i[r], mt);
            alpha[r] = exp2f(m_i[r] - mn);
            m_i[r] = mn;
            float p0 = exp2f(s0[r] - mn);
            float p1 = exp2f(s1[r] - mn);
#pragma unroll
            for (int j = 0; j < 4; ++j) acco[j][r] *= alpha[r];
            int rl = r + 8 * half;
            plds[wave][rl * 32 + lrow]      = (_Float16)p0;
            plds[wave][rl * 32 + 16 + lrow] = (_Float16)p1;
        }
        // D-layout -> A-layout re-tile through wave-private LDS; same-wave
        // LDS ordering makes this safe without a workgroup barrier.
        wave_lds_fence();
        h16v ap = h16_combine(*(const h8v*)&plds[wave][lrow * 32 + half * 16],
                              *(const h8v*)&plds[wave][lrow * 32 + 8 + half * 16]);
        wave_lds_fence();

        // ---- row sums via P @ ones (broadcast across the half in D layout)
        f8v lz = {};
        f8v lsum = wmma_f16(ap, ones, lz);

        // ---- O += P V ----
#pragma unroll
        for (int j = 0; j < 4; ++j)
            acco[j] = wmma_f16(ap, vf[j], acco[j]);

#pragma unroll
        for (int r = 0; r < 8; ++r)
            l_i[r] = l_i[r] * alpha[r] + lsum[r];
    }

    // ---- normalize + merge heads into [B, N, DIM] f16 ----
#pragma unroll
    for (int j = 0; j < 4; ++j) {
#pragma unroll
        for (int r = 0; r < 8; ++r) {
            int row = q0 + r + 8 * half;
            int c = h * HD + 16 * j + lrow;
            float val = acco[j][r] / l_i[r];
            O[((size_t)b * NSEQ + row) * DIM + c] = (_Float16)val;
        }
    }
}

// ---------------------------------------------------------------------------
extern "C" void kernel_launch(void* const* d_in, const int* in_sizes, int n_in,
                              void* d_out, int out_size, void* d_ws, size_t ws_size,
                              hipStream_t stream) {
    (void)in_sizes; (void)n_in; (void)out_size; (void)ws_size;
    const float* q     = (const float*)d_in[0];
    const float* k     = (const float*)d_in[1];
    const float* v     = (const float*)d_in[2];
    const float* gq    = (const float*)d_in[3];
    const float* bq_ln = (const float*)d_in[4];
    const float* gk    = (const float*)d_in[5];
    const float* bk_ln = (const float*)d_in[6];
    const float* gv    = (const float*)d_in[7];
    const float* bv_ln = (const float*)d_in[8];
    const float* Wq    = (const float*)d_in[9];
    const float* bq    = (const float*)d_in[10];
    const float* Wk    = (const float*)d_in[11];
    const float* bk    = (const float*)d_in[12];
    const float* Wv    = (const float*)d_in[13];
    const float* bv    = (const float*)d_in[14];
    const float* Wp    = (const float*)d_in[15];
    const float* bp    = (const float*)d_in[16];

    float* out = (float*)d_out;                                   // [B,N,DIM]
    float* kh  = out + (size_t)BATCH * NSEQ * DIM;                // [B,H,N,hd]
    float* vh  = kh  + (size_t)BATCH * HEADS * NSEQ * HD;         // [B,H,N,hd]

    char* ws = (char*)d_ws;
    // f16 scratch layout (bytes); total 56 MiB
    _Float16* qn16 = (_Float16*)(ws + 0);          // 8 MiB  LN(q)
    _Float16* kn16 = (_Float16*)(ws + 8388608);    // 8 MiB  LN(k)
    _Float16* vn16 = (_Float16*)(ws + 16777216);   // 8 MiB  LN(v)
    _Float16* wq_t = (_Float16*)(ws + 25165824);   // 2 MiB  Wq^T
    _Float16* wk_t = (_Float16*)(ws + 27262976);   // 2 MiB  Wk^T
    _Float16* wv_t = (_Float16*)(ws + 29360128);   // 2 MiB  Wv^T
    _Float16* wp_t = (_Float16*)(ws + 31457280);   // 2 MiB  Wp^T
    _Float16* q16  = (_Float16*)(ws + 33554432);   // 8 MiB  qh  [B,H,N,hd] (pre-scaled)
    _Float16* k16  = (_Float16*)(ws + 41943040);   // 8 MiB  kh  [B,H,N,hd]
    _Float16* vt16 = (_Float16*)(ws + 50331648);   // 8 MiB  vh  [B,H,hd,N]
    _Float16* a16  = qn16;  // attention output aliases qn16 (dead by then)

    ln_kernel<<<dim3(ROWS, 3), 256, 0, stream>>>(
        q, k, v, gq, bq_ln, gk, bk_ln, gv, bv_ln, qn16, kn16, vn16);

    cvtw_kernel<<<dim3(DIM / 32, DIM / 32, 4), dim3(32, 8), 0, stream>>>(
        Wq, Wk, Wv, Wp, wq_t, wk_t, wv_t, wp_t);

    dim3 ggrid(ROWS / 256, DIM / 64);
    gemm_kernel<0, 1, true ><<<ggrid, 256, 0, stream>>>(qn16, wq_t, bq, nullptr, q16);
    gemm_kernel<2, 1, false><<<ggrid, 256, 0, stream>>>(kn16, wk_t, bk, kh,      k16);
    gemm_kernel<2, 2, false><<<ggrid, 256, 0, stream>>>(vn16, wv_t, bv, vh,      vt16);

    attn_kernel<<<dim3(NSEQ / 128, HEADS, BATCH), 256, 0, stream>>>(
        q16, k16, vt16, a16);

    gemm_kernel<1, 0, false><<<ggrid, 256, 0, stream>>>(a16, wp_t, bp, out, nullptr);
}
